// NasLSTM_76682346103408
// MI455X (gfx1250) — matrix operand order repaired and checked by reading, not compiled, not run on hardware
//
#include <hip/hip_runtime.h>
#include <hip/hip_bf16.h>
#include <math.h>

// ---------------- problem constants ----------------
#define H        2048            // hidden
#define G4       8192            // 4*H gate rows
#define T_STEPS  48
#define S_OPT    8
#define NWG      64              // persistent grid: 64 WGs x 256 thr = 512 waves
#define TPB      256
#define KTILES   (H / 32)        // 64 k-chunks of 32 per 16-row tile

typedef __bf16 bf16_t;
typedef __bf16 v16bf __attribute__((ext_vector_type(16)));
typedef float  v8f   __attribute__((ext_vector_type(8)));

union Frag16 { v16bf v; uint4 q[2]; };

// ---------------- helpers ----------------
__device__ __forceinline__ float sigmoidf_(float v) { return 1.0f / (1.0f + expf(-v)); }

__device__ __forceinline__ unsigned mix32(unsigned x) {
  x ^= x >> 17; x *= 0xED5AD4BBu;
  x ^= x >> 11; x *= 0xAC4C1B51u;
  x ^= x >> 15; x *= 0x31848BABu;
  x ^= x >> 14;
  return x;
}

// device-scope sense-reversal grid barrier (grid is small enough to be co-resident)
__device__ __forceinline__ void grid_barrier(unsigned* cnt, unsigned* gen, unsigned nwg) {
  __syncthreads();
  if (threadIdx.x == 0) {
    __threadfence();  // make prior global stores visible device-wide
    unsigned g = __hip_atomic_load(gen, __ATOMIC_RELAXED, __HIP_MEMORY_SCOPE_AGENT);
    unsigned arrived =
        __hip_atomic_fetch_add(cnt, 1u, __ATOMIC_ACQ_REL, __HIP_MEMORY_SCOPE_AGENT);
    if (arrived == nwg - 1u) {
      __hip_atomic_store(cnt, 0u, __ATOMIC_RELAXED, __HIP_MEMORY_SCOPE_AGENT);
      __hip_atomic_store(gen, g + 1u, __ATOMIC_RELEASE, __HIP_MEMORY_SCOPE_AGENT);
    } else {
      while (__hip_atomic_load(gen, __ATOMIC_ACQUIRE, __HIP_MEMORY_SCOPE_AGENT) == g) {
        __builtin_amdgcn_s_sleep(2);
      }
    }
    __threadfence();
  }
  __syncthreads();
}

// ---------------- workspace init: zero first 64KB (h, c, x, hbf, barrier state) ----
__global__ void init_ws(unsigned* ws32, unsigned nwords) {
  unsigned stride = blockDim.x * gridDim.x;
  for (unsigned j = blockIdx.x * blockDim.x + threadIdx.x; j < nwords; j += stride)
    ws32[j] = 0u;
}

// ---------------- W_hh f32 -> bf16, pre-swizzled into WMMA A-fragment order -------
// Layout: tile rt (16 rows), k-iter i (32 k's), lane l -> 16 contiguous bf16 (32B)
// so the matvec wave reads one contiguous, fully coalesced 1KB block per k-iter.
// A-fragment (16-bit A 16x32, contiguous k): lane m   : K = k0..k0+15
//                                           lane m+16: K = k0+16..k0+31
__global__ void convert_swizzle(const float* __restrict__ W, bf16_t* __restrict__ Wbf) {
  unsigned gid = blockIdx.x * blockDim.x + threadIdx.x;  // 512*64*32 = 1,048,576 threads
  if (gid >= (unsigned)(G4 / 16) * KTILES * 32) return;
  unsigned rt   = gid >> 11;        // / (64*32)
  unsigned rem  = gid & 2047u;
  unsigned i    = rem >> 5;
  unsigned l    = rem & 31u;
  unsigned mrow = l & 15u, half = l >> 4;
  unsigned row  = rt * 16u + mrow;
  unsigned k0   = i * 32u + half * 16u;
  const float* src0 = W + (size_t)row * H + k0;
  const float* src1 = src0 + 8;
  bf16_t* dst = Wbf + (size_t)rt * (16 * H) + i * 512u + l * 16u;
#pragma unroll
  for (int j = 0; j < 8; ++j) dst[j] = (bf16_t)src0[j];
#pragma unroll
  for (int j = 0; j < 8; ++j) dst[8 + j] = (bf16_t)src1[j];
}

// ---------------- persistent controller kernel ----------------
__global__ void nas_lstm_persistent(
    const float* __restrict__ W_ih, const float* __restrict__ b_ih,
    const float* __restrict__ W_hh, const float* __restrict__ b_hh,
    const float* __restrict__ head_W, const float* __restrict__ head_b,
    float* __restrict__ out,
    float* gates, float* h, float* c, float* xbuf,
    unsigned* cnt, unsigned* gen,
    bf16_t* hbf, const bf16_t* Wbf, int useWmma) {

  __shared__ float s_logits[S_OPT];

  const int tid  = threadIdx.x;
  const int lane = tid & 31;
  const int wid  = (blockIdx.x * TPB + tid) >> 5;  // 0..511, one 16-row tile each

  // Broadcast-B base: lanes 0-15 read h[k0..k0+15], lanes 16-31 read h[k0+16..k0+31].
  // All 16 B columns carry the same h chunk -> D columns are replicated; no EXEC
  // masking or fragment zeroing needed in the inner loop.
  const bf16_t* bbase = hbf + (lane >> 4) * 16;

  for (int t = 0; t < T_STEPS; ++t) {
    const float x = *(volatile float*)xbuf;  // sample fed back from previous step

    // -------- phase 1: gate pre-activations, 16 rows per wave --------
    if (useWmma) {
      const bf16_t* wt = Wbf + (size_t)wid * (16 * H);
      v8f acc = {0.f, 0.f, 0.f, 0.f, 0.f, 0.f, 0.f, 0.f};
      for (int i = 0; i < KTILES; ++i) {
        Frag16 a, b;
        const uint4* pa = (const uint4*)(wt + i * 512 + lane * 16);
        a.q[0] = pa[0];
        a.q[1] = pa[1];
        const uint4* pb = (const uint4*)(bbase + i * 32);  // broadcast 64B/wave
        b.q[0] = pb[0];
        b.q[1] = pb[1];
        acc = __builtin_amdgcn_wmma_f32_16x16x32_bf16(
            false, a.v, false, b.v, (short)0, acc, false, false);
      }
      if ((lane & 15) == 0) {  // D column 0: lane0 -> M=0..7, lane16 -> M=8..15
        int rbase = wid * 16 + (lane >> 4) * 8;
#pragma unroll
        for (int r = 0; r < 8; ++r) {
          int row = rbase + r;
          gates[row] = acc[r] + W_ih[row] * x + b_ih[row] + b_hh[row];
        }
      }
    } else {  // fallback: coalesced f32 FMA matvec (ws too small for bf16 copy)
      for (int rr = 0; rr < 16; ++rr) {
        int row = wid * 16 + rr;
        const float* wr = W_hh + (size_t)row * H;
        float acc = 0.f;
        for (int k = lane; k < H; k += 32) acc += wr[k] * h[k];
        for (int off = 16; off; off >>= 1) acc += __shfl_down(acc, off, 32);
        if (lane == 0) gates[row] = acc + W_ih[row] * x + b_ih[row] + b_hh[row];
      }
    }

    grid_barrier(cnt, gen, NWG);

    // -------- phase 2 (WG0 only): LSTM pointwise, head matvec, sample --------
    if (blockIdx.x == 0) {
      for (int j = tid; j < H; j += TPB) {
        float ig = sigmoidf_(gates[j]);
        float fg = sigmoidf_(gates[H + j]);
        float gg = tanhf(gates[2 * H + j]);
        float og = sigmoidf_(gates[3 * H + j]);
        float cv = fg * c[j] + ig * gg;
        c[j] = cv;
        float hv = og * tanhf(cv);
        h[j] = hv;
        hbf[j] = (bf16_t)hv;
      }
      __syncthreads();

      // head: wave s computes logit s (8 waves, wave32 shuffle reduction)
      int w = tid >> 5;
      const float* hw = head_W + ((size_t)t * S_OPT + w) * H;
      float acc = 0.f;
      for (int k = lane; k < H; k += 32) acc += hw[k] * h[k];
      for (int off = 16; off; off >>= 1) acc += __shfl_down(acc, off, 32);
      if (lane == 0) s_logits[w] = acc + head_b[t * S_OPT + w];
      __syncthreads();

      if (tid == 0) {
        float m = s_logits[0];
        for (int s = 1; s < S_OPT; ++s) m = fmaxf(m, s_logits[s]);
        float e[S_OPT], sum = 0.f;
        for (int s = 0; s < S_OPT; ++s) { e[s] = expf(s_logits[s] - m); sum += e[s]; }
        // categorical sample = argmax(logits + Gumbel), deterministic hash RNG
        int best = 0; float bz = -1e30f;
        for (int s = 0; s < S_OPT; ++s) {
          unsigned r = mix32(0x9E3779B9u ^ (unsigned)(t * S_OPT + s + 1));
          float u = ((float)r + 0.5f) * (1.0f / 4294967296.0f);
          float z = s_logits[s] - logf(-logf(u) + 1e-20f);
          if (z > bz) { bz = z; best = s; }
        }
        out[t] = e[best] / sum;       // selected action probability
        *xbuf = (float)best;          // feedback input for next step
      }
    }

    grid_barrier(cnt, gen, NWG);
  }
}

// ---------------- launcher ----------------
extern "C" void kernel_launch(void* const* d_in, const int* in_sizes, int n_in,
                              void* d_out, int out_size, void* d_ws, size_t ws_size,
                              hipStream_t stream) {
  const float* W_ih   = (const float*)d_in[0];
  const float* b_ih   = (const float*)d_in[1];
  const float* W_hh   = (const float*)d_in[2];
  const float* b_hh   = (const float*)d_in[3];
  const float* head_W = (const float*)d_in[4];
  const float* head_b = (const float*)d_in[5];
  float* out = (float*)d_out;

  char* base = (char*)d_ws;
  float*    gates = (float*)base;              // 8192 f32
  float*    h     = gates + G4;                // 2048 f32
  float*    c     = h + H;                     // 2048 f32
  float*    xbuf  = c + H;                     // 1 f32 (+pad)
  unsigned* cnt   = (unsigned*)(xbuf + 4);
  unsigned* gen   = cnt + 1;
  bf16_t*   hbf   = (bf16_t*)(base + 50 * 1024);  // 4KB, 32B aligned
  bf16_t*   Wbf   = (bf16_t*)(base + 64 * 1024);  // 32MB swizzled bf16 W_hh

  const size_t small_bytes = 64 * 1024;
  const size_t need_wmma   = small_bytes + (size_t)G4 * H * sizeof(bf16_t);
  const int useWmma = (ws_size >= need_wmma) ? 1 : 0;

  unsigned init_words = (unsigned)((ws_size < small_bytes ? ws_size : small_bytes) / 4);
  hipLaunchKernelGGL(init_ws, dim3(16), dim3(TPB), 0, stream, (unsigned*)base, init_words);

  if (useWmma) {
    // 512 tiles * 64 k-iters * 32 lanes = 1,048,576 threads
    hipLaunchKernelGGL(convert_swizzle, dim3(4096), dim3(TPB), 0, stream, W_hh, Wbf);
  }

  hipLaunchKernelGGL(nas_lstm_persistent, dim3(NWG), dim3(TPB), 0, stream,
                     W_ih, b_ih, W_hh, b_hh, head_W, head_b, out,
                     gates, h, c, xbuf, cnt, gen, hbf, Wbf, useWmma);
}
